// LSTMClassifier_86792699118254
// MI455X (gfx1250) — matrix operand order repaired and checked by reading, not compiled, not run on hardware
//
#include <hip/hip_runtime.h>

// ---------------------------------------------------------------------------
// Problem constants (from reference): B=64, T=256, E=512, H=1024, 4H=4096, L=10
// ---------------------------------------------------------------------------
constexpr int BB = 64;
constexpr int TT = 256;
constexpr int EE = 512;
constexpr int HH = 1024;
constexpr int GG = 4 * HH;   // 4096
constexpr int LL = 10;

typedef __bf16 bf16_t;
typedef __attribute__((ext_vector_type(16))) __bf16 v16bf;
typedef __attribute__((ext_vector_type(8)))  __bf16 v8bf;
typedef __attribute__((ext_vector_type(8)))  float  v8f;

#define CAT16(lo, hi) __builtin_shufflevector(lo, hi, 0,1,2,3,4,5,6,7,8,9,10,11,12,13,14,15)

__device__ __forceinline__ bf16_t f2bf(float x) {
    // round-to-nearest-even fp32 -> bf16
    unsigned u = __builtin_bit_cast(unsigned, x);
    unsigned r = u + 0x7FFFu + ((u >> 16) & 1u);
    return __builtin_bit_cast(bf16_t, (unsigned short)(r >> 16));
}

__device__ __forceinline__ float sigmoidf_(float x) {
    return 1.0f / (1.0f + __expf(-x));
}

// ---------------------------------------------------------------------------
// Prep kernels
// ---------------------------------------------------------------------------

// W [K][N] fp32 row-major  ->  WT [N][K] bf16 row-major (transpose + convert)
__global__ void conv_transpose_kernel(const float* __restrict__ W,
                                      bf16_t* __restrict__ WT,
                                      int K, int N) {
    size_t idx = (size_t)blockIdx.x * blockDim.x + threadIdx.x;
    size_t total = (size_t)K * N;
    if (idx >= total) return;
    int k = (int)(idx / N);
    int n = (int)(idx % N);
    WT[(size_t)n * K + k] = f2bf(W[idx]);
}

// Xb[t][b][e] = bf16(emb[sentence[b][t]][e]); one thread per 4 elements
__global__ void gather_embed_kernel(const int* __restrict__ sent,
                                    const float* __restrict__ emb,
                                    bf16_t* __restrict__ Xb) {
    size_t idx = (size_t)blockIdx.x * blockDim.x + threadIdx.x;   // per float4
    size_t total = (size_t)TT * BB * (EE / 4);
    if (idx >= total) return;
    int e4   = (int)(idx % (EE / 4));
    size_t r = idx / (EE / 4);
    int b    = (int)(r % BB);
    int t    = (int)(r / BB);
    int tok  = sent[b * TT + t];
    const float4 v = *(const float4*)(emb + (size_t)tok * EE + (size_t)e4 * 4);
    bf16_t* dst = Xb + ((size_t)t * BB + b) * EE + (size_t)e4 * 4;
    dst[0] = f2bf(v.x); dst[1] = f2bf(v.y); dst[2] = f2bf(v.z); dst[3] = f2bf(v.w);
}

__global__ void zero_f32_kernel(float* __restrict__ p, int n) {
    int i = blockIdx.x * blockDim.x + threadIdx.x;
    if (i < n) p[i] = 0.0f;
}

__global__ void zero_bf16_kernel(bf16_t* __restrict__ p, int n) {
    int i = blockIdx.x * blockDim.x + threadIdx.x;
    if (i < n) p[i] = f2bf(0.0f);
}

// ---------------------------------------------------------------------------
// One K-phase of the gate GEMM, 2-stage software pipelined.
//   A: 4 M-tiles (rows mt*16 + lane&15), row stride K
//   B: one gate-column row of WT, contiguous K
//   next-iteration loads are issued BEFORE current WMMAs so the scheduler can
//   overlap global_load_b128 latency with v_wmma via partial s_wait_loadcnt.
// ---------------------------------------------------------------------------
template<int K>
__device__ __forceinline__ void gemm_phase(const bf16_t* __restrict__ aRow,
                                           const bf16_t* __restrict__ bBase,
                                           const int hiA, const int hiB,
                                           v8f acc[4]) {
    v8bf b0 = *(const v8bf*)(bBase + hiB);
    v8bf b1 = *(const v8bf*)(bBase + hiB + 8);
    v8bf a0[4], a1[4];
    #pragma unroll
    for (int mt = 0; mt < 4; ++mt) {
        const bf16_t* ap = aRow + (size_t)(mt * 16) * K;
        a0[mt] = *(const v8bf*)(ap + hiA);
        a1[mt] = *(const v8bf*)(ap + 8 + hiA);
    }
    #pragma unroll 2
    for (int kk = 0; kk < K - 32; kk += 32) {
        const int kn = kk + 32;
        // stage: issue next iteration's loads first
        v8bf nb0 = *(const v8bf*)(bBase + kn + hiB);
        v8bf nb1 = *(const v8bf*)(bBase + kn + hiB + 8);
        v8bf na0[4], na1[4];
        #pragma unroll
        for (int mt = 0; mt < 4; ++mt) {
            const bf16_t* ap = aRow + (size_t)(mt * 16) * K + kn;
            na0[mt] = *(const v8bf*)(ap + hiA);
            na1[mt] = *(const v8bf*)(ap + 8 + hiA);
        }
        // compute on current fragments
        v16bf bv = CAT16(b0, b1);
        #pragma unroll
        for (int mt = 0; mt < 4; ++mt) {
            v16bf av = CAT16(a0[mt], a1[mt]);
            acc[mt] = __builtin_amdgcn_wmma_f32_16x16x32_bf16(false, av, false, bv,
                                                              (short)0, acc[mt], false, false);
        }
        // rotate
        b0 = nb0; b1 = nb1;
        #pragma unroll
        for (int mt = 0; mt < 4; ++mt) { a0[mt] = na0[mt]; a1[mt] = na1[mt]; }
    }
    // epilogue
    v16bf bv = CAT16(b0, b1);
    #pragma unroll
    for (int mt = 0; mt < 4; ++mt) {
        v16bf av = CAT16(a0[mt], a1[mt]);
        acc[mt] = __builtin_amdgcn_wmma_f32_16x16x32_bf16(false, av, false, bv,
                                                          (short)0, acc[mt], false, false);
    }
}

// ---------------------------------------------------------------------------
// LSTM step: gates = Xt @ Wx + h @ Wh + bias ; apply i/f/g/o ; update c, h
//   grid = 64 WGs (one per 16 hidden cols); block = 128 threads = 4 waves
//   wave w computes gate w for ALL 4 M-tiles: each weight fragment is loaded
//   once per step (12 MB/step total from L2) and feeds 4 WMMAs.
// ---------------------------------------------------------------------------
__global__ __launch_bounds__(128)
void lstm_step_kernel(const bf16_t* __restrict__ Xt,    // [B][E] bf16 (slice t)
                      const bf16_t* __restrict__ WxT,   // [4H][E] bf16
                      const bf16_t* __restrict__ WhT,   // [4H][H] bf16
                      const float*  __restrict__ bias,  // [4H]
                      const bf16_t* __restrict__ hin,   // [B][H] bf16
                      bf16_t*       __restrict__ hout,  // [B][H] bf16
                      float*        __restrict__ c,     // [B][H] f32 (in-place)
                      float*        __restrict__ hf)    // [B][H] f32
{
    const int wave = threadIdx.x >> 5;     // gate index: 0=i 1=f 2=g 3=o
    const int lane = threadIdx.x & 31;
    const int hbk  = blockIdx.x;           // hidden col block 0..63

    const int ln   = lane & 15;
    const int hiA  = (lane >> 4) * 16;     // A-frag K sub-offset per lane half
    const int hiB  = (lane >> 4) * 16;     // B-frag K sub-offset per lane half
    const int ncol = wave * HH + hbk * 16 + ln;   // gate column in [0,4096)

    v8f acc[4] = {};   // one 16x16 accumulator per M-tile (batch rows 16*mt..)

    // phase 1: x_t @ Wx (K=512); phase 2: h @ Wh (K=1024)
    gemm_phase<EE>(Xt  + (size_t)ln * EE, WxT + (size_t)ncol * EE, hiA, hiB, acc);
    gemm_phase<HH>(hin + (size_t)ln * HH, WhT + (size_t)ncol * HH, hiA, hiB, acc);

    // ---- bias + activation, stash gate tiles in LDS ----
    // C/D layout per tile: VGPR r -> (M = mt*16 + r + 8*(lane>=16), N = lane&15)
    __shared__ float gt[4][BB][17];
    {
        const float bv  = bias[ncol];
        const int mbase = (lane >> 4) * 8;
        #pragma unroll
        for (int mt = 0; mt < 4; ++mt) {
            #pragma unroll
            for (int r = 0; r < 8; ++r) {
                float x = acc[mt][r] + bv;
                gt[wave][mt * 16 + mbase + r][ln] = (wave == 2) ? tanhf(x) : sigmoidf_(x);
            }
        }
    }
    __syncthreads();

    // ---- combine gates: c' = f*c + i*g ; h = o*tanh(c')  (64 rows x 16 cols) ----
    for (int e = threadIdx.x; e < BB * 16; e += 128) {
        int row = e >> 4;                 // batch row 0..63
        int Nl  = e & 15;
        float iv = gt[0][row][Nl];
        float fv = gt[1][row][Nl];
        float gv = gt[2][row][Nl];
        float ov = gt[3][row][Nl];
        int col = hbk * 16 + Nl;
        size_t ci = (size_t)row * HH + col;
        float cn = fv * c[ci] + iv * gv;
        c[ci] = cn;
        float hv = ov * tanhf(cn);
        hf[ci]   = hv;
        hout[ci] = f2bf(hv);
    }
}

// ---------------------------------------------------------------------------
// Final classifier: out[b][l] = h[b] . Wl[:,l] + bl[l]   (64x10, negligible)
// ---------------------------------------------------------------------------
__global__ void final_linear_kernel(const float* __restrict__ hf,
                                    const float* __restrict__ Wl,
                                    const float* __restrict__ bl,
                                    float* __restrict__ out) {
    int tid = blockIdx.x * blockDim.x + threadIdx.x;
    if (tid >= BB * LL) return;
    int b = tid / LL, l = tid % LL;
    float s = bl[l];
    for (int k = 0; k < HH; ++k)
        s += hf[(size_t)b * HH + k] * Wl[(size_t)k * LL + l];
    out[tid] = s;
}

// ---------------------------------------------------------------------------
// Host-side orchestration (all on `stream`; graph-capture safe)
// ---------------------------------------------------------------------------
extern "C" void kernel_launch(void* const* d_in, const int* in_sizes, int n_in,
                              void* d_out, int out_size, void* d_ws, size_t ws_size,
                              hipStream_t stream) {
    const int*   sent = (const int*)  d_in[0];   // [B][T]
    const float* emb  = (const float*)d_in[1];   // [V][E]
    const float* Wx   = (const float*)d_in[2];   // [E][4H]
    const float* Wh   = (const float*)d_in[3];   // [H][4H]
    const float* bias = (const float*)d_in[4];   // [4H]
    const float* Wl   = (const float*)d_in[5];   // [H][L]
    const float* bl   = (const float*)d_in[6];   // [L]
    float* out = (float*)d_out;                  // [B][L]

    // Workspace layout (~29 MB total)
    char* ws = (char*)d_ws;
    bf16_t* WxT = (bf16_t*)(ws);                              //  4 MiB [4096][512]
    bf16_t* WhT = (bf16_t*)(ws + (4u  << 20));                //  8 MiB [4096][1024]
    bf16_t* Xb  = (bf16_t*)(ws + (12u << 20));                // 16 MiB [T][B][E]
    float*  c   = (float*) (ws + (28u << 20));                // 256 KiB [B][H]
    float*  hf  = (float*) (ws + (28u << 20) + 262144u);      // 256 KiB [B][H]
    bf16_t* hb0 = (bf16_t*)(ws + (28u << 20) + 524288u);      // 128 KiB [B][H]
    bf16_t* hb1 = (bf16_t*)(ws + (28u << 20) + 524288u + 131072u);

    // prep: convert/transpose weights, gather embeddings, zero state
    conv_transpose_kernel<<<(EE * GG + 255) / 256, 256, 0, stream>>>(Wx, WxT, EE, GG);
    conv_transpose_kernel<<<(HH * GG + 255) / 256, 256, 0, stream>>>(Wh, WhT, HH, GG);
    gather_embed_kernel<<<((size_t)TT * BB * (EE / 4) + 255) / 256, 256, 0, stream>>>(sent, emb, Xb);
    zero_f32_kernel <<<(BB * HH + 255) / 256, 256, 0, stream>>>(c, BB * HH);
    zero_bf16_kernel<<<(BB * HH + 255) / 256, 256, 0, stream>>>(hb0, BB * HH);

    // recurrence: one launch per timestep, ping-pong bf16 hidden state
    for (int t = 0; t < TT; ++t) {
        const bf16_t* Xt   = Xb + (size_t)t * BB * EE;
        const bf16_t* hin  = (t & 1) ? hb1 : hb0;
        bf16_t*       hout = (t & 1) ? hb0 : hb1;
        lstm_step_kernel<<<64, 128, 0, stream>>>(Xt, WxT, WhT, bias, hin, hout, c, hf);
    }

    // classifier head
    final_linear_kernel<<<(BB * LL + 255) / 256, 256, 0, stream>>>(hf, Wl, bl, out);
}
